// mLSTMLayer_40063454937569
// MI455X (gfx1250) — matrix operand order repaired and checked by reading, not compiled, not run on hardware
//
#include <hip/hip_runtime.h>
#include <hip/hip_bf16.h>

#define B_   2
#define S_   1024
#define D_   1024
#define NH_  8
#define QK_  512
#define V_   1024
#define DQK_ 64
#define DHV_ 128

typedef __attribute__((ext_vector_type(16))) __bf16 v16bf;
typedef __attribute__((ext_vector_type(8)))  float  v8f;

// ---------------------------------------------------------------------------
// fp32 -> bf16 conversion
// ---------------------------------------------------------------------------
__global__ void f32_to_bf16_kernel(const float* __restrict__ in,
                                   __bf16* __restrict__ out, int n) {
  int i = blockIdx.x * blockDim.x + threadIdx.x;
  if (i < n) out[i] = (__bf16)in[i];
}

// ---------------------------------------------------------------------------
// WMMA GEMM: C[M,N] (f32) = A[M,K] (bf16, row-major) * Bw[N,K]^T (bf16, rm)
// grid = (N/64, M/128), block = 128 threads (4 waves).
// Wave w computes rows [by*128 + w*32, +32), cols [bx*64, +64):
//   2 A fragments x 4 B fragments -> 8 WMMA tiles, 8 v_wmma per K=32 step.
// Fragment layouts follow CDNA5 ISA (wave32):
//   A 16x32 bf16 : lane m=lane&15, kbase=(lane>>4)*8 ; halves 0-7 -> K=kb+0..7,
//                  halves 8-15 -> K=16+kb+0..7
//   B 32x16 bf16 : lane n=lane&15, kbase=(lane>>4)*8 ; same K order as A
//   C/D 16x16 f32: lane n=lane&15, mbase=(lane>>4)*8 ; vgpr r -> M=mbase+r
// ---------------------------------------------------------------------------
__global__ void gemm_bf16_nt(const __bf16* __restrict__ A,
                             const __bf16* __restrict__ Bw,
                             float* __restrict__ C, int M, int N, int K) {
  const int lane = threadIdx.x & 31;
  const int wave = threadIdx.x >> 5;
  const int row0 = blockIdx.y * 128 + wave * 32;
  const int col0 = blockIdx.x * 64;

  const int am  = lane & 15;
  const int akb = (lane >> 4) * 8;
  const int bn  = lane & 15;
  const int bkb = (lane >> 4) * 8;

  const __bf16* ap0 = A + (size_t)(row0 + am) * K;
  const __bf16* ap1 = ap0 + (size_t)16 * K;
  const __bf16* bp0 = Bw + (size_t)(col0 + bn) * K + bkb;

  v8f zero = {};
  v8f acc[8];
#pragma unroll
  for (int t = 0; t < 8; ++t) acc[t] = zero;

  for (int k0 = 0; k0 < K; k0 += 32) {
    __builtin_prefetch(ap0 + k0 + 128, 0, 1);   // global_prefetch_b8
    __builtin_prefetch(ap1 + k0 + 128, 0, 1);

    v16bf a0, a1;
#pragma unroll
    for (int i = 0; i < 8; ++i) {
      a0[i]     = ap0[k0 + akb + i];
      a0[8 + i] = ap0[k0 + 16 + akb + i];
      a1[i]     = ap1[k0 + akb + i];
      a1[8 + i] = ap1[k0 + 16 + akb + i];
    }

#pragma unroll
    for (int t = 0; t < 4; ++t) {
      const __bf16* bp = bp0 + (size_t)t * 16 * K + k0;
      v16bf b;
#pragma unroll
      for (int i = 0; i < 8; ++i) { b[i] = bp[i]; b[8 + i] = bp[16 + i]; }
      acc[t] = __builtin_amdgcn_wmma_f32_16x16x32_bf16(
          false, a0, false, b, (short)0, acc[t], false, false);
      acc[t + 4] = __builtin_amdgcn_wmma_f32_16x16x32_bf16(
          false, a1, false, b, (short)0, acc[t + 4], false, false);
    }
  }

  const int cn  = lane & 15;
  const int cmb = (lane >> 4) * 8;
#pragma unroll
  for (int t = 0; t < 4; ++t) {
    float* cp0 = C + (size_t)(row0 + cmb) * N + col0 + t * 16 + cn;
    float* cp1 = cp0 + (size_t)16 * N;
#pragma unroll
    for (int r = 0; r < 8; ++r) {
      cp0[(size_t)r * N] = acc[t][r];
      cp1[(size_t)r * N] = acc[t + 4][r];
    }
  }
}

// ---------------------------------------------------------------------------
// Gate pre-activations: i_pre/f_pre = soft_cap(x @ W^T + b, 15)
// one block per row; x row staged in LDS; 16 threads do the 16 dots.
// ---------------------------------------------------------------------------
__global__ void gates_kernel(const float* __restrict__ x,
                             const float* __restrict__ Wi, const float* __restrict__ bi,
                             const float* __restrict__ Wf, const float* __restrict__ bf,
                             float* __restrict__ ipre, float* __restrict__ fpre) {
  __shared__ float sx[D_];
  const int row = blockIdx.x;
  for (int d = threadIdx.x; d < D_; d += blockDim.x)
    sx[d] = x[(size_t)row * D_ + d];
  __syncthreads();
  const int t = threadIdx.x;
  if (t < 16) {
    const int h = t & 7;
    const float* w = (t < 8) ? (Wi + (size_t)h * D_) : (Wf + (size_t)h * D_);
    float s = (t < 8) ? bi[h] : bf[h];
    for (int d = 0; d < D_; ++d) s += sx[d] * w[d];
    const float c = 15.0f * tanhf(s * (1.0f / 15.0f));
    if (t < 8) ipre[row * NH_ + h] = c;
    else       fpre[row * NH_ + h] = c;
  }
}

// ---------------------------------------------------------------------------
// Exact max-stabilized mLSTM recurrence. grid = B*NH blocks, block = 128.
// Thread j owns column j of C (64 f32 in registers). q/k for the NEXT
// timestep are prefetched into a double-buffered LDS slot with CDNA5 async
// global->LDS copies (ASYNCcnt-tracked); s_wait_asynccnt 2 keeps the newest
// prefetch in flight while guaranteeing the current buffer is complete.
// q-scale (DQK^-0.5) is applied at the end (h and qn are linear in q).
// ---------------------------------------------------------------------------
__global__ void recurrent_kernel(const float* __restrict__ q,  // [B*S, QK]
                                 const float* __restrict__ k,  // [B*S, QK]
                                 const float* __restrict__ v,  // [B*S, V]
                                 const float* __restrict__ ipre,
                                 const float* __restrict__ fpre,
                                 float* __restrict__ hout) {   // [B*S, V]
  const int b  = blockIdx.x >> 3;
  const int hh = blockIdx.x & 7;
  const int j  = threadIdx.x;   // 0..127

  __shared__ float sqk[2][2][DQK_];   // [buf][0=q,1=k][d]
  __shared__ float sn[DQK_], sred[DQK_];
  __shared__ float sqn_s;

  float C[DQK_];
#pragma unroll
  for (int d = 0; d < DQK_; ++d) C[d] = 0.0f;
  if (j < DQK_) sn[j] = 0.0f;

  const size_t qkbase = (size_t)(b * S_) * QK_ + hh * DQK_;
  // synchronous fill of buffer 0 (t = 0)
  if (j < DQK_) {
    sqk[0][0][j] = q[qkbase + j];
    sqk[0][1][j] = k[qkbase + j];
  }

  float m = 0.0f;
  const float scale = 0.125f;  // DQK^-0.5

  for (int t = 0; t < S_; ++t) {
    const int row = b * S_ + t;
    const float ip = ipre[row * NH_ + hh];
    const float fp = fpre[row * NH_ + hh];
    const float flog = (fp >= 0.0f) ? -log1pf(expf(-fp)) : (fp - log1pf(expf(fp)));
    const float m_new = fmaxf(flog + m, ip);
    const float f_act = expf(flog + m - m_new);
    const float i_act = expf(ip - m_new);
    m = m_new;

    if (j < DQK_) {   // wave-uniform: waves 0,1 enter; waves 2,3 skip
      if (t + 1 < S_) {
        const float* gq = q + qkbase + (size_t)(t + 1) * QK_ + j;
        const float* gk = k + qkbase + (size_t)(t + 1) * QK_ + j;
        const unsigned lq = (unsigned)(uintptr_t)&sqk[(t + 1) & 1][0][j];
        const unsigned lk = (unsigned)(uintptr_t)&sqk[(t + 1) & 1][1][j];
        asm volatile("global_load_async_to_lds_b32 %0, %1, off"
                     :: "v"(lq), "v"(gq) : "memory");
        asm volatile("global_load_async_to_lds_b32 %0, %1, off"
                     :: "v"(lk), "v"(gk) : "memory");
        // wait until only the two just-issued prefetches remain outstanding
        asm volatile("s_wait_asynccnt 0x2" ::: "memory");
      } else {
        asm volatile("s_wait_asynccnt 0x0" ::: "memory");
      }
    }
    __syncthreads();  // buffer t complete and visible to all waves

    const float* sq = sqk[t & 1][0];
    const float* sk = sqk[t & 1][1];

    if (j < DQK_) {
      const float nd = f_act * sn[j] + i_act * sk[j];
      sn[j] = nd;
      sred[j] = sq[j] * nd;   // raw q; scaled after reduction
    }

    const float vj  = v[(size_t)row * V_ + hh * DHV_ + j];
    const float ikv = i_act * vj;
    float acc = 0.0f;
#pragma unroll
    for (int d = 0; d < DQK_; ++d) {
      C[d] = f_act * C[d] + ikv * sk[d];
      acc += sq[d] * C[d];
    }
    __syncthreads();  // sred ready; all reads of buffer t done
    if (j == 0) {
      float s = 0.0f;
      for (int d = 0; d < DQK_; ++d) s += sred[d];
      sqn_s = s;
    }
    __syncthreads();  // sqn ready

    const float qn = sqn_s * scale;
    const float denom = fmaxf(fabsf(qn), expf(-m_new)) + 1e-6f;
    hout[(size_t)row * V_ + hh * DHV_ + j] = (acc * scale) / denom;
  }
}

// ---------------------------------------------------------------------------
// Per-(row, head) layernorm over DHV, * gamma, gated by sigmoid(o_pre) -> bf16
// grid = (B*S)*NH blocks, block = 128.
// ---------------------------------------------------------------------------
__global__ void norm_gate_kernel(const float* __restrict__ h,
                                 const float* __restrict__ opre,
                                 const float* __restrict__ gamma,
                                 __bf16* __restrict__ out) {
  const int bid = blockIdx.x;
  const int row = bid >> 3;
  const int hh  = bid & 7;
  const int j   = threadIdx.x;   // 0..127

  __shared__ float red[DHV_];
  __shared__ float s_mu, s_rstd;

  const size_t base = (size_t)row * V_ + hh * DHV_;
  const float val = h[base + j];
  red[j] = val;
  __syncthreads();
  if (j == 0) {
    float s = 0.0f;
    for (int i = 0; i < DHV_; ++i) s += red[i];
    s_mu = s * (1.0f / DHV_);
  }
  __syncthreads();
  const float dv = val - s_mu;
  red[j] = dv * dv;
  __syncthreads();
  if (j == 0) {
    float s = 0.0f;
    for (int i = 0; i < DHV_; ++i) s += red[i];
    s_rstd = rsqrtf(s * (1.0f / DHV_) + 1e-6f);
  }
  __syncthreads();
  const float hn = dv * s_rstd * gamma[hh * DHV_ + j];
  const float o  = opre[base + j];
  const float g  = 1.0f / (1.0f + expf(-o));
  out[base + j] = (__bf16)(g * hn);
}

// ---------------------------------------------------------------------------
// Launcher
// ---------------------------------------------------------------------------
extern "C" void kernel_launch(void* const* d_in, const int* in_sizes, int n_in,
                              void* d_out, int out_size, void* d_ws, size_t ws_size,
                              hipStream_t stream) {
  const float* x     = (const float*)d_in[0];
  const float* Wq    = (const float*)d_in[1];
  const float* Wk    = (const float*)d_in[2];
  const float* Wv    = (const float*)d_in[3];
  const float* Wog   = (const float*)d_in[4];
  const float* Wi    = (const float*)d_in[5];
  const float* bi    = (const float*)d_in[6];
  const float* Wf    = (const float*)d_in[7];
  const float* bfv   = (const float*)d_in[8];
  const float* gamma = (const float*)d_in[9];
  const float* Wout  = (const float*)d_in[10];
  float* y = (float*)d_out;

  const int ROWS = B_ * S_;  // 2048

  size_t off = 0;
  auto alloc = [&](size_t bytes) -> char* {
    char* p = (char*)d_ws + off;
    off += (bytes + 255) & ~(size_t)255;
    return p;
  };
  __bf16* x_bf    = (__bf16*)alloc((size_t)ROWS * D_ * 2);
  __bf16* wq_bf   = (__bf16*)alloc((size_t)QK_ * D_ * 2);
  __bf16* wk_bf   = (__bf16*)alloc((size_t)QK_ * D_ * 2);
  __bf16* wv_bf   = (__bf16*)alloc((size_t)V_ * D_ * 2);
  __bf16* wog_bf  = (__bf16*)alloc((size_t)V_ * D_ * 2);
  __bf16* wout_bf = (__bf16*)alloc((size_t)D_ * V_ * 2);
  float*  qbuf    = (float*)alloc((size_t)ROWS * QK_ * 4);
  float*  kbuf    = (float*)alloc((size_t)ROWS * QK_ * 4);
  float*  vbuf    = (float*)alloc((size_t)ROWS * V_ * 4);
  float*  obuf    = (float*)alloc((size_t)ROWS * V_ * 4);
  float*  ibuf    = (float*)alloc((size_t)ROWS * NH_ * 4);
  float*  fbuf    = (float*)alloc((size_t)ROWS * NH_ * 4);
  float*  hbuf    = (float*)alloc((size_t)ROWS * V_ * 4);
  __bf16* hg_bf   = (__bf16*)alloc((size_t)ROWS * V_ * 2);
  (void)ws_size; (void)n_in; (void)in_sizes; (void)out_size;

  auto cvt = [&](const float* src, __bf16* dst, int n) {
    f32_to_bf16_kernel<<<(n + 255) / 256, 256, 0, stream>>>(src, dst, n);
  };
  cvt(x,    x_bf,    ROWS * D_);
  cvt(Wq,   wq_bf,   QK_ * D_);
  cvt(Wk,   wk_bf,   QK_ * D_);
  cvt(Wv,   wv_bf,   V_ * D_);
  cvt(Wog,  wog_bf,  V_ * D_);
  cvt(Wout, wout_bf, D_ * V_);

  // projections: out = x @ W^T   (grid = N/64 x M/128)
  gemm_bf16_nt<<<dim3(QK_ / 64, ROWS / 128), 128, 0, stream>>>(x_bf, wq_bf,  qbuf, ROWS, QK_, D_);
  gemm_bf16_nt<<<dim3(QK_ / 64, ROWS / 128), 128, 0, stream>>>(x_bf, wk_bf,  kbuf, ROWS, QK_, D_);
  gemm_bf16_nt<<<dim3(V_  / 64, ROWS / 128), 128, 0, stream>>>(x_bf, wv_bf,  vbuf, ROWS, V_,  D_);
  gemm_bf16_nt<<<dim3(V_  / 64, ROWS / 128), 128, 0, stream>>>(x_bf, wog_bf, obuf, ROWS, V_,  D_);

  gates_kernel<<<ROWS, 256, 0, stream>>>(x, Wi, bi, Wf, bfv, ibuf, fbuf);

  recurrent_kernel<<<B_ * NH_, 128, 0, stream>>>(qbuf, kbuf, vbuf, ibuf, fbuf, hbuf);

  norm_gate_kernel<<<ROWS * NH_, 128, 0, stream>>>(hbuf, obuf, gamma, hg_bf);

  // y = h_out @ Wout^T
  gemm_bf16_nt<<<dim3(D_ / 64, ROWS / 128), 128, 0, stream>>>(hg_bf, wout_bf, y, ROWS, D_, V_);
}
